// MRConv4dTF_35493609734668
// MI455X (gfx1250) — matrix operand order repaired and checked by reading, not compiled, not run on hardware
//
#include <hip/hip_runtime.h>
#include <hip/hip_bf16.h>

typedef __attribute__((ext_vector_type(16))) __bf16 v16bf;
typedef __attribute__((ext_vector_type(8)))  float  v8f;
typedef __attribute__((ext_vector_type(4)))  int    v4i;

#define BATCH 8
#define CIN   96
#define NTOK  3136
#define COUT  192
#define KDIM  192        // 2*CIN
#define BN_EPS 1e-5f

#define AS1 __attribute__((address_space(1)))
#define AS3 __attribute__((address_space(3)))

// Async global->LDS copy support (guarded; falls back to batched VGPR copies)
#if __has_builtin(__builtin_amdgcn_global_load_async_to_lds_b128)
#define ASYNC_MODE 1
#elif __has_builtin(__builtin_amdgcn_load_to_lds)
#define ASYNC_MODE 2
#else
#define ASYNC_MODE 0
#endif

// Flat pointer -> address-space-qualified pointers expected by the builtin.
// Flat LDS addr = {aperture, offset[31:0]} (ISA 10.2), so truncation yields
// the LDS offset; global flat addresses are numerically identical in AS1.
__device__ __forceinline__ AS1 v4i* as_global_v4i(const void* p) {
  return (AS1 v4i*)(unsigned long long)p;
}
__device__ __forceinline__ AS3 v4i* as_lds_v4i(void* p) {
  return (AS3 v4i*)(unsigned int)(unsigned long long)p;
}

__device__ __forceinline__ unsigned short f2bf_bits(float f) {
  unsigned u = __builtin_bit_cast(unsigned, f);
  unsigned r = u + 0x7FFFu + ((u >> 16) & 1u);   // round-to-nearest-even
  return (unsigned short)(r >> 16);
}
__device__ __forceinline__ __bf16 bfh(unsigned short b) {
  return __builtin_bit_cast(__bf16, b);
}

// ---- DPP 16-lane rotate-reduce (VALU only, no DS, no waits) ---------------
template <int CTRL>
__device__ __forceinline__ float dpp_ror(float x) {
  return __builtin_bit_cast(
      float, __builtin_amdgcn_update_dpp(0, __builtin_bit_cast(int, x), CTRL,
                                         0xF, 0xF, true));
}
__device__ __forceinline__ float row_max16(float x) {
  x = fmaxf(x, dpp_ror<0x121>(x));
  x = fmaxf(x, dpp_ror<0x122>(x));
  x = fmaxf(x, dpp_ror<0x124>(x));
  x = fmaxf(x, dpp_ror<0x128>(x));
  return x;
}
__device__ __forceinline__ float row_sum16(float x) {
  x += dpp_ror<0x121>(x);
  x += dpp_ror<0x122>(x);
  x += dpp_ror<0x124>(x);
  x += dpp_ror<0x128>(x);
  return x;
}

// 16 contiguous bf16 halves -> B-operand / generic v16bf
__device__ __forceinline__ v16bf load16(const unsigned short* p) {
  union { float4 f[2]; v16bf v; } u;
  u.f[0] = *(const float4*)(p);
  u.f[1] = *(const float4*)(p + 8);
  return u.v;
}
// two contiguous 8-half runs -> A-operand v16bf (documented 16x32 A layout)
__device__ __forceinline__ v16bf load8x2(const unsigned short* p0, const unsigned short* p1) {
  union { float4 f[2]; v16bf v; } u;
  u.f[0] = *(const float4*)(p0);
  u.f[1] = *(const float4*)(p1);
  return u.v;
}

__device__ __forceinline__ v8f wmma_bf16(v16bf a, v16bf b, v8f c) {
  // D = A(16x32) * B(32x16) + C, f32 accumulate
  return __builtin_amdgcn_wmma_f32_16x16x32_bf16(false, a, false, b, (short)0, c, false, false);
}

// ---------------------------------------------------------------------------
// Kernel 0: zero BN stat accumulators (sum, sumsq per output channel)
// ---------------------------------------------------------------------------
__global__ void zero_stats_kernel(float* __restrict__ stats) {
  int i = threadIdx.x;
  if (i < 2 * COUT) stats[i] = 0.0f;
}

// ---------------------------------------------------------------------------
// Kernel 1: pack x f32 -> bf16 in BOTH layouts:
//   qbf  [B,N,C] token-major  (A-operands / score B-operands)
//   qbfT [B,C,N] channel-major (V B-operands; same layout as x)
// ---------------------------------------------------------------------------
__global__ void pack_q_kernel(const float* __restrict__ x,
                              unsigned short* __restrict__ qbf,
                              unsigned short* __restrict__ qbfT) {
  size_t idx = (size_t)blockIdx.x * blockDim.x + threadIdx.x;
  size_t total = (size_t)BATCH * CIN * NTOK;
  if (idx >= total) return;
  size_t n = idx % NTOK;
  size_t c = (idx / NTOK) % CIN;
  size_t b = idx / ((size_t)NTOK * CIN);
  unsigned short h = f2bf_bits(x[idx]);
  qbfT[idx] = h;                       // channel-major (x layout)
  qbf[(b * NTOK + n) * CIN + c] = h;   // token-major
}

// ---------------------------------------------------------------------------
// Kernel 2: flash attention (q == k == v), online softmax, bf16 WMMA, f32 acc
//   block = 128 threads = 4 waves; each wave owns one 16-query tile
//   loop over 98 chunks of 32 keys; kv (token-major) and kvT (channel-major)
//   staged by async global->LDS copies when available.
// ---------------------------------------------------------------------------
__global__ __launch_bounds__(128) void attn_kernel(const unsigned short* __restrict__ qbf,
                                                   const unsigned short* __restrict__ qbfT,
                                                   float* __restrict__ o) {
  __shared__ __align__(16) unsigned short kv[32][CIN];        // [key][chan]
  __shared__ __align__(16) unsigned short kvT[CIN][32];       // [chan][key]
  __shared__ __align__(16) unsigned short pstage[4][16][32];  // per-wave P tile

  const int tid  = threadIdx.x;
  const int wave = tid >> 5;
  const int lane = tid & 31;
  const int hi   = lane >> 4;      // half-wave select (VGPR layout)
  const int col  = lane & 15;      // matrix column / A-row index
  const int b    = blockIdx.y;
  const int q0   = (blockIdx.x * 4 + wave) * 16;

  // Q tile A-operands: 3 chunks of K=32 covering C=96
  const unsigned short* qrow = qbf + ((size_t)b * NTOK + q0 + col) * CIN;
  v16bf qA[3];
#pragma unroll
  for (int ck = 0; ck < 3; ++ck)
    qA[ck] = load16(qrow + ck * 32 + 16 * hi);

  v8f zero = {};
  v8f oacc[6];
#pragma unroll
  for (int t = 0; t < 6; ++t) oacc[t] = zero;
  float m[8], l[8];
#pragma unroll
  for (int v = 0; v < 8; ++v) { m[v] = -1e30f; l[v] = 0.0f; }

  const unsigned short* kbase  = qbf  + (size_t)b * NTOK * CIN;  // token-major
  const unsigned short* kbaseT = qbfT + (size_t)b * CIN * NTOK;  // channel-major

  for (int kc = 0; kc < NTOK / 32; ++kc) {
    __syncthreads();
    const unsigned short* src = kbase + (size_t)kc * 32 * CIN;

    // prefetch next chunk (global_prefetch_b8) while this one computes
    if (kc + 1 < NTOK / 32) {
      __builtin_prefetch((const char*)(src + 32 * CIN) + tid * 48, 0, 3);
      if (tid < CIN)
        __builtin_prefetch(kbaseT + (size_t)tid * NTOK + (kc + 1) * 32, 0, 3);
    }

    // six 16-byte staging slots per thread (3 for kv, 3 for kvT)
    const unsigned short* gsrc[6];
    unsigned short*       ldst[6];
#pragma unroll
    for (int i = 0; i < 3; ++i) {
      int idx = tid + i * 128;            // 0..383
      int row = idx / 12;                 // key 0..31
      int c0  = (idx % 12) * 8;           // channel base
      gsrc[i] = src + row * CIN + c0;
      ldst[i] = &kv[row][c0];
      int c = idx >> 2;                   // channel 0..95
      int j = idx & 3;                    // 8-half block in row
      gsrc[3 + i] = kbaseT + (size_t)c * NTOK + kc * 32 + j * 8;
      ldst[3 + i] = &kvT[c][j * 8];
    }
#if ASYNC_MODE == 1
#pragma unroll
    for (int s = 0; s < 6; ++s)
      __builtin_amdgcn_global_load_async_to_lds_b128(as_global_v4i(gsrc[s]),
                                                     as_lds_v4i(ldst[s]), 0, 0);
    asm volatile("s_wait_asynccnt 0x0" ::: "memory");
#elif ASYNC_MODE == 2
#pragma unroll
    for (int s = 0; s < 6; ++s)
      __builtin_amdgcn_load_to_lds((void*)gsrc[s], (AS3 void*)as_lds_v4i(ldst[s]), 16, 0, 0);
    asm volatile("s_wait_asynccnt 0x0\n\ts_wait_loadcnt 0x0" ::: "memory");
#else
    {
      float4 tmp[6];
#pragma unroll
      for (int s = 0; s < 6; ++s) tmp[s] = *(const float4*)gsrc[s];
#pragma unroll
      for (int s = 0; s < 6; ++s) *(float4*)ldst[s] = tmp[s];
    }
#endif
    __syncthreads();

    // scores: S0 = Q*K^T (keys 0..15), S1 = keys 16..31; K dim = channels.
    v16bf sb[6];
#pragma unroll
    for (int ck = 0; ck < 3; ++ck) {
      sb[2 * ck]     = load16(&kv[col][ck * 32 + 16 * hi]);
      sb[2 * ck + 1] = load16(&kv[16 + col][ck * 32 + 16 * hi]);
    }
    v8f s0 = {}, s1 = {};
#pragma unroll
    for (int ck = 0; ck < 3; ++ck) {
      s0 = wmma_bf16(qA[ck], sb[2 * ck], s0);
      s1 = wmma_bf16(qA[ck], sb[2 * ck + 1], s1);
    }

    // online softmax: rows live in (vgpr v, lane-half hi); cols in lanes 0..15
#pragma unroll
    for (int v = 0; v < 8; ++v) {
      float mx = row_max16(fmaxf(s0[v], s1[v]));
      float mn   = fmaxf(m[v], mx);
      float corr = __expf(m[v] - mn);
      m[v] = mn;
      float p0 = __expf(s0[v] - mn);
      float p1 = __expf(s1[v] - mn);
      s0[v] = p0; s1[v] = p1;
      float rs = row_sum16(p0 + p1);
      l[v] = l[v] * corr + rs;
#pragma unroll
      for (int t = 0; t < 6; ++t) oacc[t][v] *= corr;
    }

    // V B-operands (independent of P staging) — issue loads early
    v16bf vb[6];
#pragma unroll
    for (int t = 0; t < 6; ++t)
      vb[t] = load16(&kvT[t * 16 + col][16 * hi]);

    // P (C-layout f32) -> bf16 A-operand via per-wave LDS bounce
    unsigned short* ps = &pstage[wave][0][0];
#pragma unroll
    for (int v = 0; v < 8; ++v) {
      ps[(v + 8 * hi) * 32 + col]      = f2bf_bits(s0[v]);
      ps[(v + 8 * hi) * 32 + 16 + col] = f2bf_bits(s1[v]);
    }
    asm volatile("s_wait_dscnt 0" ::: "memory");   // wave-local LDS RAW
    v16bf pA = load16(ps + col * 32 + 16 * hi);

    // O += P(16x32) * V(32x16 per channel chunk); back-to-back WMMAs
#pragma unroll
    for (int t = 0; t < 6; ++t)
      oacc[t] = wmma_bf16(pA, vb[t], oacc[t]);
  }

  // normalize and store O [B,N,C] f32
  float* orow = o + ((size_t)b * NTOK + q0) * CIN;
#pragma unroll
  for (int v = 0; v < 8; ++v) {
    float inv = 1.0f / l[v];
#pragma unroll
    for (int t = 0; t < 6; ++t)
      orow[(size_t)(v + 8 * hi) * CIN + t * 16 + col] = oacc[t][v] * inv;
  }
}

// ---------------------------------------------------------------------------
// Kernel 3: 1x1 conv over cat=[x, relu(x - att)] + bias, write y into d_out
// ---------------------------------------------------------------------------
__global__ __launch_bounds__(128) void conv_bn_stats_kernel(
    const float* __restrict__ x, const float* __restrict__ o,
    const float* __restrict__ conv_w, const float* __restrict__ conv_b,
    float* __restrict__ y, float* __restrict__ stats) {
  __shared__ __align__(16) unsigned short catT[64][KDIM];  // [token][in-chan]

  const int tid  = threadIdx.x;
  const int wave = tid >> 5;
  const int lane = tid & 31;
  const int hi   = lane >> 4;
  const int col  = lane & 15;
  const int b    = blockIdx.z;
  const int n0   = blockIdx.x * 64;
  const int o0   = blockIdx.y * 64;
  const int om   = o0 + wave * 16;

  // build transposed cat tile in bf16
  const float* xb = x + (size_t)b * CIN * NTOK;
  const float* ob = o + (size_t)b * NTOK * CIN;
  for (int i = tid; i < CIN * 64; i += 128) {
    int c = i / 64;
    int t = i % 64;
    float xv = xb[(size_t)c * NTOK + n0 + t];
    float ov = ob[(size_t)(n0 + t) * CIN + c];
    float xj = fmaxf(xv - ov, 0.0f);
    catT[t][c]       = f2bf_bits(xv);
    catT[t][CIN + c] = f2bf_bits(xj);
  }
  __syncthreads();

  v8f zero = {};
  v8f acc[4] = {zero, zero, zero, zero};
#pragma unroll
  for (int kc = 0; kc < 6; ++kc) {
    // A = conv_w rows [om..om+15] x K-chunk (gather f32 -> bf16)
    const float* wr = conv_w + (size_t)(om + col) * KDIM + kc * 32;
    v16bf a;
#pragma unroll
    for (int j = 0; j < 8; ++j) a[j] = bfh(f2bf_bits(wr[8 * hi + j]));
#pragma unroll
    for (int j = 0; j < 8; ++j) a[8 + j] = bfh(f2bf_bits(wr[16 + 8 * hi + j]));
    v16bf bm[4];
#pragma unroll
    for (int ct = 0; ct < 4; ++ct)
      bm[ct] = load8x2(&catT[ct * 16 + col][kc * 32 + 8 * hi], &catT[ct * 16 + col][kc * 32 + 16 + 8 * hi]);
#pragma unroll
    for (int ct = 0; ct < 4; ++ct)
      acc[ct] = wmma_bf16(a, bm[ct], acc[ct]);
  }

  // bias, store, per-channel stats (rows = out channels)
#pragma unroll
  for (int v = 0; v < 8; ++v) {
    int ch = om + v + 8 * hi;
    float bias = conv_b[ch];
    float s = 0.0f, qq = 0.0f;
#pragma unroll
    for (int ct = 0; ct < 4; ++ct) {
      float yv = acc[ct][v] + bias;
      y[((size_t)b * COUT + ch) * NTOK + n0 + ct * 16 + col] = yv;
      s += yv;
      qq += yv * yv;
    }
    s  = row_sum16(s);
    qq = row_sum16(qq);
    if (col == 0) {
      atomicAdd(&stats[ch], s);
      atomicAdd(&stats[COUT + ch], qq);
    }
  }
}

// ---------------------------------------------------------------------------
// Kernel 4: BatchNorm (training-mode biased stats) + exact erf GELU, in place
// ---------------------------------------------------------------------------
__global__ void bn_gelu_kernel(float* __restrict__ y, const float* __restrict__ stats,
                               const float* __restrict__ gamma, const float* __restrict__ beta) {
  size_t idx = (size_t)blockIdx.x * blockDim.x + threadIdx.x;
  size_t total = (size_t)BATCH * COUT * NTOK;
  if (idx >= total) return;
  int ch = (int)((idx / NTOK) % COUT);
  const float invCnt = 1.0f / (float)(BATCH * NTOK);
  float mean = stats[ch] * invCnt;
  float var  = stats[COUT + ch] * invCnt - mean * mean;
  float r = rsqrtf(var + BN_EPS);
  float yn = (y[idx] - mean) * r * gamma[ch] + beta[ch];
  y[idx] = 0.5f * yn * (1.0f + erff(yn * 0.70710678118654752f));
}

// ---------------------------------------------------------------------------
extern "C" void kernel_launch(void* const* d_in, const int* in_sizes, int n_in,
                              void* d_out, int out_size, void* d_ws, size_t ws_size,
                              hipStream_t stream) {
  const float* x      = (const float*)d_in[0];
  const float* conv_w = (const float*)d_in[1];
  const float* conv_b = (const float*)d_in[2];
  const float* gamma  = (const float*)d_in[3];
  const float* beta   = (const float*)d_in[4];
  float* y = (float*)d_out;

  char* w = (char*)d_ws;
  size_t half_count = (size_t)BATCH * NTOK * CIN;
  unsigned short* qbf  = (unsigned short*)w;                       // 4.6 MB token-major bf16
  unsigned short* qbfT = (unsigned short*)(w + half_count * 2);    // 4.6 MB channel-major bf16
  float* o = (float*)(w + half_count * 4);                         // 9.6 MB f32 att out
  float* stats = (float*)(w + half_count * 4 + half_count * 4);    // 2*192 floats

  zero_stats_kernel<<<1, 2 * COUT, 0, stream>>>(stats);

  {
    size_t total = (size_t)BATCH * CIN * NTOK;
    pack_q_kernel<<<(unsigned)((total + 255) / 256), 256, 0, stream>>>(x, qbf, qbfT);
  }

  attn_kernel<<<dim3(NTOK / 64, BATCH), 128, 0, stream>>>(qbf, qbfT, o);

  conv_bn_stats_kernel<<<dim3(NTOK / 64, COUT / 64, BATCH), 128, 0, stream>>>(
      x, o, conv_w, conv_b, y, stats);

  {
    size_t total = (size_t)BATCH * COUT * NTOK;
    bn_gelu_kernel<<<(unsigned)((total + 255) / 256), 256, 0, stream>>>(y, stats, gamma, beta);
  }
}
